// Appnp_16286515986694
// MI455X (gfx1250) — matrix operand order repaired and checked by reading, not compiled, not run on hardware
//
#include <hip/hip_runtime.h>

#define IN_F   512
#define HID_F  256
#define OUT_F  16
#define ALPHA  0.1f
#define KITER  10
#define NT1    8     // n-tiles per wave in GEMM1 (8*16 = 128 output cols per wave)

typedef __attribute__((ext_vector_type(16))) __bf16         v16bf;
typedef __attribute__((ext_vector_type(8)))  __bf16         v8bf;
typedef __attribute__((ext_vector_type(8)))  float          v8f;
typedef __attribute__((ext_vector_type(4)))  unsigned short ushort4v;

static __device__ __forceinline__ unsigned short f32_to_bf16_bits(float f) {
  unsigned int u = __builtin_bit_cast(unsigned int, f);
  u += 0x7FFFu + ((u >> 16) & 1u);            // round-to-nearest-even
  return (unsigned short)(u >> 16);
}

static __device__ __forceinline__ v16bf cat16(v8bf lo, v8bf hi) {
  return __builtin_shufflevector(lo, hi, 0, 1, 2, 3, 4, 5, 6, 7,
                                         8, 9, 10, 11, 12, 13, 14, 15);
}

// ---------------------------------------------------------------------------
// Elementwise fp32 -> bf16 (RNE). total must be a multiple of 4 (true for all
// our arrays: dims are multiples of 256/512). Pure bandwidth, runs once.
// ---------------------------------------------------------------------------
__global__ __launch_bounds__(256) void cvt_f32_bf16(const float* __restrict__ in,
                                                    unsigned short* __restrict__ out,
                                                    int total) {
  const int base = (blockIdx.x * blockDim.x + threadIdx.x) * 4;
  if (base >= total) return;
  const float4 f = *(const float4*)(in + base);
  ushort4v u;
  u.x = f32_to_bf16_bits(f.x);
  u.y = f32_to_bf16_bits(f.y);
  u.z = f32_to_bf16_bits(f.z);
  u.w = f32_to_bf16_bits(f.w);
  *(ushort4v*)(out + base) = u;
}

// ---------------------------------------------------------------------------
// GEMM1: H1 = relu(Xb @ W1b^T + b1), pure bf16 WMMA, fp32 accumulate.
// One wave owns one 16-row m-tile and NT1 consecutive 16-col n-tiles:
// the A fragment is loaded once per k-step and reused for NT1 WMMAs.
// A (16x32 bf16): lane L -> row m0+L%16, K chunks at koffA and koffA+8.
// B (32x16 bf16): lane L -> col n +L%16, 16 contiguous K at koffB.
// ---------------------------------------------------------------------------
__global__ __launch_bounds__(256) void gemm1_relu_wmma(
    const unsigned short* __restrict__ Xb, const unsigned short* __restrict__ W1b,
    const float* __restrict__ b1, unsigned short* __restrict__ H1,
    int Nn, int MT) {
  const int lane   = threadIdx.x & 31;
  const int wave   = threadIdx.x >> 5;
  const int m_tile = blockIdx.x * 8 + wave;
  if (m_tile >= MT) return;                 // uniform per-wave: EXEC stays all-1
  const int n0   = blockIdx.y * (NT1 * 16); // 0 or 128
  const int m0   = m_tile << 4;
  const int half = lane >> 4;               // 0 or 1
  const int l15  = lane & 15;
  int mrow = m0 + l15; if (mrow >= Nn) mrow = Nn - 1;   // clamp (loads only)
  const __bf16* xrow = (const __bf16*)Xb  + (size_t)mrow * IN_F;
  const __bf16* wbase = (const __bf16*)W1b;
  const int koffA = half * 16;
  const int koffB = half * 16;

  v8f acc[NT1] = {};
#pragma unroll
  for (int kt = 0; kt < IN_F / 32; ++kt) {
    const int k0 = kt * 32;
    const v8bf a0 = *(const v8bf*)(xrow + k0 + koffA);
    const v8bf a1 = *(const v8bf*)(xrow + k0 + koffA + 8);
    const v16bf a = cat16(a0, a1);
#pragma unroll
    for (int nt = 0; nt < NT1; ++nt) {
      const __bf16* wrow = wbase + (size_t)(n0 + nt * 16 + l15) * IN_F;
      const v16bf b = *(const v16bf*)(wrow + k0 + koffB);
      acc[nt] = __builtin_amdgcn_wmma_f32_16x16x32_bf16(false, a, false, b,
                                                        (short)0, acc[nt],
                                                        false, false);
    }
  }
  // C/D layout: VGPR r -> M = r + 8*half, N = l15
  const int mbase = m0 + half * 8;
#pragma unroll
  for (int nt = 0; nt < NT1; ++nt) {
    const int nj = n0 + nt * 16 + l15;
    const float bias = b1[nj];
#pragma unroll
    for (int r = 0; r < 8; ++r) {
      const int m = mbase + r;
      if (m < Nn) {
        float v = acc[nt][r] + bias;
        v = v > 0.f ? v : 0.f;
        H1[(size_t)m * HID_F + nj] = f32_to_bf16_bits(v);
      }
    }
  }
}

// ---------------------------------------------------------------------------
// GEMM2: h0 = H1 @ W2b^T + b2  (OUT=16 -> single N tile). Also seeds z0 = h0.
// ---------------------------------------------------------------------------
__global__ __launch_bounds__(256) void gemm2_wmma(
    const unsigned short* __restrict__ H1, const unsigned short* __restrict__ W2b,
    const float* __restrict__ b2, float* __restrict__ h0,
    float* __restrict__ z0, int Nn, int MT) {
  const int lane   = threadIdx.x & 31;
  const int wave   = threadIdx.x >> 5;
  const int m_tile = blockIdx.x * 8 + wave;
  if (m_tile >= MT) return;
  const int m0   = m_tile << 4;
  const int half = lane >> 4;
  const int l15  = lane & 15;
  int mrow = m0 + l15; if (mrow >= Nn) mrow = Nn - 1;
  const __bf16* h1row = (const __bf16*)H1  + (size_t)mrow * HID_F;
  const __bf16* wrow  = (const __bf16*)W2b + (size_t)l15  * HID_F;
  const int koffA = half * 16;
  const int koffB = half * 16;

  v8f c = {};
#pragma unroll
  for (int kt = 0; kt < HID_F / 32; ++kt) {
    const int k0 = kt * 32;
    const v8bf a0 = *(const v8bf*)(h1row + k0 + koffA);
    const v8bf a1 = *(const v8bf*)(h1row + k0 + koffA + 8);
    const v16bf a = cat16(a0, a1);
    const v16bf b = *(const v16bf*)(wrow + k0 + koffB);
    c = __builtin_amdgcn_wmma_f32_16x16x32_bf16(false, a, false, b,
                                                (short)0, c, false, false);
  }
  const float bias  = b2[l15];
  const int   mbase = m0 + half * 8;
#pragma unroll
  for (int r = 0; r < 8; ++r) {
    const int m = mbase + r;
    if (m < Nn) {
      const float v = c[r] + bias;
      const size_t o = (size_t)m * OUT_F + l15;
      h0[o] = v;
      z0[o] = v;
    }
  }
}

// ---------------------------------------------------------------------------
// Graph normalization
// ---------------------------------------------------------------------------
__global__ __launch_bounds__(256) void init_deg(float* __restrict__ deg, int Nn) {
  int i = blockIdx.x * blockDim.x + threadIdx.x;
  if (i < Nn) deg[i] = 1.0f;                       // self loop
}
__global__ __launch_bounds__(256) void deg_acc(const int* __restrict__ col,
                                               float* __restrict__ deg, int E) {
  int e = blockIdx.x * blockDim.x + threadIdx.x;
  if (e < E) unsafeAtomicAdd(&deg[col[e]], 1.0f);
}
__global__ __launch_bounds__(256) void dinv_selfco(const float* __restrict__ deg,
                                                   float* __restrict__ dinv,
                                                   float* __restrict__ selfco,
                                                   int Nn) {
  int i = blockIdx.x * blockDim.x + threadIdx.x;
  if (i >= Nn) return;
  const float d  = deg[i];
  const float di = d > 0.f ? rsqrtf(d) : 0.f;
  dinv[i]   = di;
  selfco[i] = (1.0f - ALPHA) * di * di;            // 0.9 * self-loop norm
}
__global__ __launch_bounds__(256) void edge_norm(const int* __restrict__ row,
                                                 const int* __restrict__ col,
                                                 const float* __restrict__ dinv,
                                                 float* __restrict__ norm09, int E) {
  int e = blockIdx.x * blockDim.x + threadIdx.x;
  if (e < E) norm09[e] = (1.0f - ALPHA) * dinv[row[e]] * dinv[col[e]];
}

// ---------------------------------------------------------------------------
// APPNP iteration: zn = 0.1*h0 + selfco*zc  (dense part, fuses self loops)
// then scatter: zn[col] += 0.9*norm * zc[row]  (one thread per edge-feature)
// ---------------------------------------------------------------------------
__global__ __launch_bounds__(256) void seed_z(const float* __restrict__ h0,
                                              const float* __restrict__ selfco,
                                              const float* __restrict__ zc,
                                              float* __restrict__ zn, int total) {
  int i = blockIdx.x * blockDim.x + threadIdx.x;
  if (i < total) zn[i] = ALPHA * h0[i] + selfco[i >> 4] * zc[i];
}
__global__ __launch_bounds__(256) void scatter_edges(const int* __restrict__ row,
                                                     const int* __restrict__ col,
                                                     const float* __restrict__ norm09,
                                                     const float* __restrict__ zc,
                                                     float* __restrict__ zn, int total) {
  int t = blockIdx.x * blockDim.x + threadIdx.x;
  if (t >= total) return;
  const int e = t >> 4;
  const int f = t & 15;
  const float msg = norm09[e] * zc[(size_t)row[e] * OUT_F + f];
  unsafeAtomicAdd(&zn[(size_t)col[e] * OUT_F + f], msg);
}

// ---------------------------------------------------------------------------
// log_softmax over 16 columns, one thread per node
// ---------------------------------------------------------------------------
__global__ __launch_bounds__(256) void log_softmax_k(const float* __restrict__ z,
                                                     float* __restrict__ out, int Nn) {
  int i = blockIdx.x * blockDim.x + threadIdx.x;
  if (i >= Nn) return;
  alignas(16) float v[16];
  const float4* p = (const float4*)(z + (size_t)i * OUT_F);
  *(float4*)(v + 0)  = p[0];
  *(float4*)(v + 4)  = p[1];
  *(float4*)(v + 8)  = p[2];
  *(float4*)(v + 12) = p[3];
  float mx = v[0];
#pragma unroll
  for (int k = 1; k < 16; ++k) mx = v[k] > mx ? v[k] : mx;
  float s = 0.f;
#pragma unroll
  for (int k = 0; k < 16; ++k) s += expf(v[k] - mx);
  const float lse = mx + logf(s);
  float* o = out + (size_t)i * OUT_F;
#pragma unroll
  for (int k = 0; k < 16; ++k) o[k] = v[k] - lse;
}

// ---------------------------------------------------------------------------
extern "C" void kernel_launch(void* const* d_in, const int* in_sizes, int n_in,
                              void* d_out, int out_size, void* d_ws, size_t ws_size,
                              hipStream_t stream) {
  const float* x  = (const float*)d_in[0];
  const int*   ei = (const int*)d_in[1];
  const float* W1 = (const float*)d_in[2];
  const float* b1 = (const float*)d_in[3];
  const float* W2 = (const float*)d_in[4];
  const float* b2 = (const float*)d_in[5];

  const int Nn = in_sizes[0] / IN_F;     // 100000
  const int E  = in_sizes[1] / 2;        // 3200000
  const int* rowI = ei;                  // edge_index[0] = source
  const int* colI = ei + E;              // edge_index[1] = target
  const int MT = (Nn + 15) / 16;

  // Workspace layout (everything we read is written first; ws is poisoned).
  char* p = (char*)d_ws;
  auto alloc = [&](size_t bytes) {
    char* r = p; p += (bytes + 255) & ~(size_t)255; return r;
  };
  unsigned short* Xb     = (unsigned short*)alloc((size_t)Nn * IN_F * 2);  // 102.4 MB
  unsigned short* W1b    = (unsigned short*)alloc((size_t)HID_F * IN_F * 2);
  unsigned short* W2b    = (unsigned short*)alloc((size_t)OUT_F * HID_F * 2);
  unsigned short* H1     = (unsigned short*)alloc((size_t)Nn * HID_F * 2); // 51.2 MB
  float*          h0     = (float*)alloc((size_t)Nn * OUT_F * 4);          //  6.4 MB
  float*          zA     = (float*)alloc((size_t)Nn * OUT_F * 4);
  float*          zB     = (float*)alloc((size_t)Nn * OUT_F * 4);
  float*          degb   = (float*)alloc((size_t)Nn * 4);
  float*          dinvb  = (float*)alloc((size_t)Nn * 4);
  float*          selfco = (float*)alloc((size_t)Nn * 4);
  float*          norm09 = (float*)alloc((size_t)E * 4);                   // 12.8 MB
  (void)ws_size; (void)n_in; (void)out_size;

  const dim3 blk(256);

  // One-time bf16 downconversion of GEMM operands (bandwidth-bound, tiny).
  const int totX  = Nn * IN_F;
  const int totW1 = HID_F * IN_F;
  const int totW2 = OUT_F * HID_F;
  cvt_f32_bf16<<<(totX / 4 + 255) / 256, blk, 0, stream>>>(x, Xb, totX);
  cvt_f32_bf16<<<(totW1 / 4 + 255) / 256, blk, 0, stream>>>(W1, W1b, totW1);
  cvt_f32_bf16<<<(totW2 / 4 + 255) / 256, blk, 0, stream>>>(W2, W2b, totW2);

  dim3 g1((MT + 7) / 8, HID_F / (16 * NT1));       // y = 2
  gemm1_relu_wmma<<<g1, blk, 0, stream>>>(Xb, W1b, b1, H1, Nn, MT);
  gemm2_wmma<<<dim3((MT + 7) / 8), blk, 0, stream>>>(H1, W2b, b2, h0, zA, Nn, MT);

  init_deg<<<(Nn + 255) / 256, blk, 0, stream>>>(degb, Nn);
  deg_acc<<<(E + 255) / 256, blk, 0, stream>>>(colI, degb, E);
  dinv_selfco<<<(Nn + 255) / 256, blk, 0, stream>>>(degb, dinvb, selfco, Nn);
  edge_norm<<<(E + 255) / 256, blk, 0, stream>>>(rowI, colI, dinvb, norm09, E);

  float* zc = zA;
  float* zn = zB;
  const int totN = Nn * OUT_F;
  const int totE = E * OUT_F;
  for (int k = 0; k < KITER; ++k) {
    seed_z<<<(totN + 255) / 256, blk, 0, stream>>>(h0, selfco, zc, zn, totN);
    scatter_edges<<<(totE + 255) / 256, blk, 0, stream>>>(rowI, colI, norm09, zc, zn, totE);
    float* t = zc; zc = zn; zn = t;
  }
  log_softmax_k<<<(Nn + 255) / 256, blk, 0, stream>>>(zc, (float*)d_out, Nn);
}
